// TiLinear_acc23_55241869361744
// MI455X (gfx1250) — 8-bit matrix operand order repaired and checked by reading, compiled, not run on hardware
//
#include <hip/hip_runtime.h>
#include <stdint.h>

typedef int v8i __attribute__((ext_vector_type(8)));

static constexpr int B_ROWS = 131072;
static constexpr int K_DIM  = 256;
static constexpr int N_DIM  = 256;
static constexpr int TILE_M = 32;
static constexpr int LDS_STRIDE = 68;


__global__ __launch_bounds__(256) void prep_kernel(const float* __restrict__ w,
                                                   uint8_t* __restrict__ ws) {
    int gid = blockIdx.x * blockDim.x + threadIdx.x;
    if (gid == 0) *reinterpret_cast<int*>(ws) = 0;
    int idx = gid * 4;
    uint32_t packed = 0;
#pragma unroll
    for (int j = 0; j < 4; ++j) {
        int q = (int)w[idx + j];
        packed |= (uint32_t)(uint8_t)(int8_t)q << (8 * j);
    }
    reinterpret_cast<uint32_t*>(ws + 16)[gid] = packed;
}

__global__ __launch_bounds__(256) void gemm_kernel(const float* __restrict__ act,
                                                   uint8_t* __restrict__ ws,
                                                   float* __restrict__ out) {
    __shared__ uint32_t lA[TILE_M * LDS_STRIDE];
    __shared__ int wg_max;

    const int tid  = threadIdx.x;
    const int lane = tid & 31;
    const int wave = tid >> 5;
    const int mgrp = wave & 1;
    const int ngrp = wave >> 1;
    const size_t mblock = (size_t)blockIdx.x * TILE_M;

    if (tid == 0) wg_max = 0;

    const float4* act4 = reinterpret_cast<const float4*>(act + mblock * K_DIM);
#pragma unroll
    for (int it = 0; it < 8; ++it) {
        int f4  = it * 256 + tid;
        int row = f4 >> 6;
        int c4  = f4 & 63;
        float4 v = act4[f4];
        uint32_t p =  (uint32_t)(uint8_t)(int8_t)(int)v.x
                   | ((uint32_t)(uint8_t)(int8_t)(int)v.y << 8)
                   | ((uint32_t)(uint8_t)(int8_t)(int)v.z << 16)
                   | ((uint32_t)(uint8_t)(int8_t)(int)v.w << 24);
        lA[row * LDS_STRIDE + c4] = p;
    }
    __syncthreads();

    const int8_t* wq = reinterpret_cast<const int8_t*>(ws + 16);
    const int rowA  = mgrp * 16 + (lane & 15);
    const int kb    = (lane >> 4) * 16;
    const int khalf = (lane >> 4) * 16;
    const int ncol  = lane & 15;

    v8i acc[4];
#pragma unroll
    for (int t = 0; t < 4; ++t)
#pragma unroll
        for (int e = 0; e < 8; ++e) acc[t][e] = 0;

#pragma unroll
    for (int kk = 0; kk < K_DIM; kk += 64) {
        const uint32_t* abase = &lA[rowA * LDS_STRIDE + ((kk + kb) >> 2)];
        v8i a;
        a[0] = abase[0];  a[1] = abase[1];
        a[2] = abase[2];  a[3] = abase[3];
        a[4] = abase[8];  a[5] = abase[9];
        a[6] = abase[10]; a[7] = abase[11];

#pragma unroll
        for (int t = 0; t < 4; ++t) {
            int n = ngrp * 64 + t * 16 + ncol;
            const int8_t* wrow = wq + (size_t)n * K_DIM + kk + khalf;
            int4 b0 = *reinterpret_cast<const int4*>(wrow);
            int4 b1 = *reinterpret_cast<const int4*>(wrow + 32);
            v8i b;
            b[0] = b0.x; b[1] = b0.y; b[2] = b0.z; b[3] = b0.w;
            b[4] = b1.x; b[5] = b1.y; b[6] = b1.z; b[7] = b1.w;
            acc[t] = __builtin_amdgcn_wmma_i32_16x16x64_iu8(
                 true, a,  true, b, acc[t],
                 false,  false);
        }
    }

    int lm = 0;
    const size_t mrow0 = mblock + (size_t)(mgrp * 16 + 8 * (lane >> 4));
    const int ncol0 = ngrp * 64 + ncol;
#pragma unroll
    for (int t = 0; t < 4; ++t) {
#pragma unroll
        for (int r = 0; r < 8; ++r) {
            int v = acc[t][r];
            int av = v < 0 ? -v : v;
            lm = lm > av ? lm : av;
            out[(mrow0 + r) * N_DIM + (ncol0 + t * 16)] = (float)v;
        }
    }
    atomicMax(&wg_max, lm);
    __syncthreads();
    if (tid == 0) atomicMax(reinterpret_cast<int*>(ws), wg_max);
}

__global__ __launch_bounds__(256) void quant_kernel(float* __restrict__ out,
                                                    const uint8_t* __restrict__ ws,
                                                    const int* __restrict__ exp_in,
                                                    const int* __restrict__ weight_exp) {
    const int gmax = *reinterpret_cast<const int*>(ws);
    const float r  = (float)gmax;
    const float bw = (r > 0.0f) ? ceilf(log2f(fmaxf(r, 1.0f))) : 0.0f;
    const float shift = bw - 7.0f;
    const bool  pos   = shift > 0.0f;
    const float safe  = fmaxf(shift, 1.0f);
    const float two_s     = exp2f(safe);
    const float inv_two_s = 1.0f / two_s;
    const float inv_half  = 1.0f / exp2f(safe - 1.0f);

    size_t i4 = (size_t)blockIdx.x * blockDim.x + threadIdx.x;
    float4* o4 = reinterpret_cast<float4*>(out);
    float4 x = o4[i4];
    float vals[4] = {x.x, x.y, x.z, x.w};
#pragma unroll
    for (int j = 0; j < 4; ++j) {
        float v = vals[j];
        float rt   = floorf(v * inv_two_s);
        float prob = fabsf(v - rt * two_s);
        float dec  = floorf(prob * inv_half);
        float rounded = fminf(fmaxf(rt + dec, -127.0f), 127.0f);
        vals[j] = pos ? rounded : (float)(int8_t)(int)v;
    }
    o4[i4] = make_float4(vals[0], vals[1], vals[2], vals[3]);

    if (i4 == 0) {
        int es = exp_in[0] + weight_exp[0] + (pos ? (int)shift : 0);
        out[(size_t)B_ROWS * N_DIM] = (float)es;
    }
}

extern "C" void kernel_launch(void* const* d_in, const int* in_sizes, int n_in,
                              void* d_out, int out_size, void* d_ws, size_t ws_size,
                              hipStream_t stream) {
    const float* act        = (const float*)d_in[0];
    const int*   exp_in     = (const int*)d_in[1];
    const float* weight     = (const float*)d_in[2];
    const int*   weight_exp = (const int*)d_in[3];
    float*   out = (float*)d_out;
    uint8_t* ws  = (uint8_t*)d_ws;

    prep_kernel<<<64, 256, 0, stream>>>(weight, ws);
    gemm_kernel<<<B_ROWS / TILE_M, 256, 0, stream>>>(act, ws, out);
    quant_kernel<<<(B_ROWS * N_DIM) / (256 * 4), 256, 0, stream>>>(out, ws, exp_in, weight_exp);
}
